// PAGLattention_57913339019655
// MI455X (gfx1250) — matrix operand order repaired and checked by reading, not compiled, not run on hardware
//
#include <hip/hip_runtime.h>

typedef __attribute__((ext_vector_type(16))) __bf16 v16bf;
typedef __attribute__((ext_vector_type(8)))  __bf16 v8bf;
typedef __attribute__((ext_vector_type(8)))  float  v8f;
typedef __attribute__((ext_vector_type(4)))  unsigned int u32x4;
typedef __attribute__((ext_vector_type(8)))  int i32x8;
typedef __attribute__((ext_vector_type(4)))  int i32x4;

static __device__ __forceinline__ v16bf make_frag(v8bf lo, v8bf hi) {
  return __builtin_shufflevector(lo, hi, 0,1,2,3,4,5,6,7,8,9,10,11,12,13,14,15);
}

// ---------------------------------------------------------------------------
// TDM: DMA a [128 rows x 32 cols] bf16 tile (row stride Kc elems) from global
// into LDS, inserting 16B of padding after every 64B row -> 80B LDS rows.
// D# built per CDNA5 ISA 8.3/8.4: group0 {count=1, lds_addr, global_addr,
// type=2}; group1 {data_size=2B, pad_enable, pad_interval=16dw, pad_amount=4dw,
// tensor_dim0=Kc, tensor_dim1=M, tile_dim0=32, tile_dim1=128, d0_stride=Kc}.
// ---------------------------------------------------------------------------
static __device__ __forceinline__ void tdm_load_a_tile(unsigned lds_off,
                                                       unsigned long long ga,
                                                       int M, int Kc) {
  u32x4 g0;
  g0[0] = 1u;                                            // count=1, user mode
  g0[1] = lds_off;                                       // lds_addr [63:32]
  g0[2] = (unsigned)(ga & 0xffffffffull);                // global_addr lo
  g0[3] = (unsigned)((ga >> 32) & 0x1ffffffull) | (2u << 30); // addr hi, type=2
  i32x8 g1;
  g1[0] = (int)((1u << 16) | (1u << 20) | (3u << 22) | (3u << 25));
  //             data_size=2B  pad_en      pad_int=16dw  pad_amt=4dw
  g1[1] = (int)(((unsigned)Kc & 0xffffu) << 16);         // tensor_dim0 lo16
  g1[2] = (int)((((unsigned)Kc >> 16) & 0xffffu) |
                (((unsigned)M & 0xffffu) << 16));        // td0 hi | td1 lo
  g1[3] = (int)((((unsigned)M >> 16) & 0xffffu) | (32u << 16)); // td1 hi | tile_dim0=32
  g1[4] = (int)128u;                                     // tile_dim1=128, tile_dim2=0
  g1[5] = (int)(unsigned)Kc;                             // tensor_dim0_stride lo32
  g1[6] = 0;                                             // stride hi / dim1_stride lo
  g1[7] = 0;
  i32x4 gz = {0, 0, 0, 0};
#if defined(__clang_major__) && __clang_major__ >= 23
  i32x8 gz8 = {0, 0, 0, 0, 0, 0, 0, 0};
  __builtin_amdgcn_tensor_load_to_lds(g0, g1, gz, gz, gz8, 0);
#else
  __builtin_amdgcn_tensor_load_to_lds(g0, g1, gz, gz, 0);
#endif
}

// ---------------------------------------------------------------------------
// bf16 WMMA GEMM / 3x3-conv-as-9-shifted-GEMMs.
//   C[b][m][n] = alpha * sum_{tap,k} A[tap][m][k] * B[b][k][shift_tap(n)]
//   (+ optional f32 residual), output f32 or bf16.
// Block = 128 threads (4 waves), block tile 128x128, wave tile 64x64
// (4x4 v_wmma_f32_16x16x32_bf16 per 32-deep K slab).
// A slabs: TDM tensor_load_to_lds (double buffered, wave0 issues + waits
// TENSORcnt). B slabs: per-lane transposed staging (handles conv border/shift),
// double buffered through registers. One barrier per slab.
// ---------------------------------------------------------------------------
__global__ __launch_bounds__(128)
void gemm_conv_wmma(const __bf16* __restrict__ A, long long sAb,
                    const __bf16* __restrict__ Bm, long long sBb,
                    float* __restrict__ outF, __bf16* __restrict__ outB, long long sOb,
                    const float* __restrict__ resid, long long sRb,
                    int M, int Ntok, int Kc, int ntaps,
                    int lw, int Hh, int Ww, float alpha)
{
  // 40-element (80B) rows: 32 data + 8 pad elems; 16B-aligned fragment chunks.
  __shared__ __attribute__((aligned(16))) __bf16 bufA[2][128][40];
  __shared__ __attribute__((aligned(16))) __bf16 bufB[2][128][40];

  const int tid   = threadIdx.x;
  const int lane  = tid & 31;
  const int wave  = tid >> 5;
  const int half  = lane >> 4;
  const int l16   = lane & 15;
  const int waveM = (wave >> 1) << 6;
  const int waveN = (wave & 1) << 6;
  const int blockM = blockIdx.x << 7;
  const int blockN = blockIdx.y << 7;
  const int b      = blockIdx.z;

  const __bf16* Ab = A + (long long)b * sAb;
  const __bf16* Bb = Bm + (long long)b * sBb;

  const int tok = blockN + tid;     // token staged by this thread
  const int ty  = tok >> lw;
  const int tx  = tok & (Ww - 1);

  v8f acc[4][4];
  const v8f vzero = {0.f,0.f,0.f,0.f,0.f,0.f,0.f,0.f};
  #pragma unroll
  for (int i = 0; i < 4; ++i)
    #pragma unroll
    for (int j = 0; j < 4; ++j)
      acc[i][j] = vzero;

  const __bf16 bz = (__bf16)0.0f;
  __bf16 breg[32];

  // --- staging helpers -----------------------------------------------------
  auto issue_tdm = [&](int tp, int kc, int nb) {
    unsigned lds_off = (unsigned)(size_t)&bufA[nb][0][0];
    unsigned long long ga = (unsigned long long)(size_t)(
        Ab + (long long)tp * M * Kc + (long long)blockM * Kc + kc);
    tdm_load_a_tile(lds_off, ga, M, Kc);
  };
  auto load_b = [&](int tp, int kc) {
    const int dy = (ntaps == 9) ? (tp / 3 - 1) : 0;
    const int dx = (ntaps == 9) ? (tp % 3 - 1) : 0;
    const int sy = ty + dy, sx = tx + dx;
    const bool ok = (sy >= 0) && (sy < Hh) && (sx >= 0) && (sx < Ww);
    const long long srcoff = (long long)sy * Ww + sx;
    #pragma unroll
    for (int j = 0; j < 32; ++j) {
      __bf16 v = bz;
      if (ok) v = Bb[(long long)(kc + j) * Ntok + srcoff];
      breg[j] = v;
    }
  };
  auto store_b = [&](int nb) {
    #pragma unroll
    for (int q = 0; q < 4; ++q) {
      v8bf t;
      #pragma unroll
      for (int j = 0; j < 8; ++j) t[j] = breg[(q << 3) + j];
      *(v8bf*)&bufB[nb][tid][q << 3] = t;
    }
  };

  // --- prologue: stage slab 0 into buffer 0 --------------------------------
  if (tid < 32) issue_tdm(0, 0, 0);
  load_b(0, 0);
  store_b(0);
  if (tid < 32) __builtin_amdgcn_s_wait_tensorcnt(0);
  __syncthreads();

  // --- main loop over K slabs (taps x Kc/32), double buffered --------------
  const int S = ntaps * (Kc >> 5);
  int tap = 0, kc = 0, cur = 0;
  for (int s = 0; s < S; ++s) {
    int ntap = tap, nkc = kc + 32;
    if (nkc == Kc) { nkc = 0; ntap = tap + 1; }
    const bool have_next = (s + 1 < S);

    if (have_next) {
      if (tid < 32) issue_tdm(ntap, nkc, cur ^ 1);   // DMA next A slab
      load_b(ntap, nkc);                             // global loads, next B slab
    }

    // fragments from current buffers
    v16bf afr[4], bfr[4];
    #pragma unroll
    for (int mi = 0; mi < 4; ++mi) {
      const __bf16* ap = &bufA[cur][waveM + (mi << 4) + l16][half << 3];
      v8bf lo = *(const v8bf*)ap;
      v8bf hi = *(const v8bf*)(ap + 16);
      afr[mi] = make_frag(lo, hi);
    }
    #pragma unroll
    for (int ni = 0; ni < 4; ++ni) {
      const __bf16* bp = &bufB[cur][waveN + (ni << 4) + l16][half << 3];
      v8bf lo = *(const v8bf*)bp;
      v8bf hi = *(const v8bf*)(bp + 16);
      bfr[ni] = make_frag(lo, hi);
    }
    #pragma unroll
    for (int mi = 0; mi < 4; ++mi)
      #pragma unroll
      for (int ni = 0; ni < 4; ++ni)
        acc[mi][ni] = __builtin_amdgcn_wmma_f32_16x16x32_bf16(
            false, afr[mi], false, bfr[ni], (short)0, acc[mi][ni], false, false);

    if (have_next) {
      store_b(cur ^ 1);                              // publish next B slab
      if (tid < 32) __builtin_amdgcn_s_wait_tensorcnt(0);
    }
    __syncthreads();
    cur ^= 1; tap = ntap; kc = nkc;
  }

  // --- epilogue: C lane layout (m = r + 8*half, n = l16) per 16x16 tile ----
  const long long obase = (long long)b * sOb;
  #pragma unroll
  for (int mi = 0; mi < 4; ++mi) {
    #pragma unroll
    for (int ni = 0; ni < 4; ++ni) {
      #pragma unroll
      for (int r = 0; r < 8; ++r) {
        const int gm = blockM + waveM + (mi << 4) + (half << 3) + r;
        const int gn = blockN + waveN + (ni << 4) + l16;
        float v = acc[mi][ni][r] * alpha;
        const long long off = (long long)gm * Ntok + gn;
        if (resid) v += resid[(long long)b * sRb + off];
        if (outF) outF[obase + off] = v;
        else      outB[obase + off] = (__bf16)v;
      }
    }
  }
}

// ---------------------------------------------------------------------------
// Helper kernels
// ---------------------------------------------------------------------------
__global__ void cvt_bf16_kernel(const float* __restrict__ s, __bf16* __restrict__ d,
                                long long n) {
  long long i = (long long)blockIdx.x * blockDim.x + threadIdx.x;
  if (i < n) d[i] = (__bf16)s[i];
}

// AdaptiveAvgPool 64x64 -> 16x16 (mean over 4x4), write bf16
__global__ void pool_kernel(const float* __restrict__ e, __bf16* __restrict__ out) {
  int o = blockIdx.x * blockDim.x + threadIdx.x;   // [0, B*256*256)
  int px = o & 15, py = (o >> 4) & 15;
  int bc = o >> 8;                                 // b*256 + c
  const float* src = e + ((long long)bc * 64 + py * 4) * 64 + px * 4;
  float s = 0.f;
  #pragma unroll
  for (int yy = 0; yy < 4; ++yy)
    #pragma unroll
    for (int xx = 0; xx < 4; ++xx)
      s += src[yy * 64 + xx];
  out[o] = (__bf16)(s * 0.0625f);
}

// w[O][I][3][3] f32 -> tap-major wt[tap][O][I] bf16
__global__ void wtrans_kernel(const float* __restrict__ w, __bf16* __restrict__ out) {
  int i = blockIdx.x * blockDim.x + threadIdx.x;   // [0, 9*256*256)
  int tap = i >> 16;
  int r = i & 65535;
  int o = r >> 8, ic = r & 255;
  out[i] = (__bf16)w[((o << 8) + ic) * 9 + tap];
}

// L2-normalize each 256-token row; optional transpose (for k^T)
__global__ void l2norm_kernel(const __bf16* __restrict__ in, __bf16* __restrict__ out,
                              int transpose) {
  int row = blockIdx.x;     // b*256 + c
  int n = threadIdx.x;      // 0..255
  float v = (float)in[(long long)row * 256 + n];
  __shared__ float red[256];
  red[n] = v * v;
  __syncthreads();
  for (int s = 128; s > 0; s >>= 1) {
    if (n < s) red[n] += red[n + s];
    __syncthreads();
  }
  float r = 1.0f / fmaxf(sqrtf(red[0]), 1e-12f);
  __bf16 ov = (__bf16)(v * r);
  if (transpose) {
    int b = row >> 8, c = row & 255;
    out[(((long long)b << 8) + n) * 256 + c] = ov;   // [b][n][c]
  } else {
    out[(long long)row * 256 + n] = ov;
  }
}

// Row softmax over 256 entries, f32 in -> bf16 out
__global__ void softmax_kernel(const float* __restrict__ in, __bf16* __restrict__ out) {
  int row = blockIdx.x;
  int n = threadIdx.x;
  float v = in[(long long)row * 256 + n];
  __shared__ float red[256];
  red[n] = v;
  __syncthreads();
  for (int s = 128; s > 0; s >>= 1) {
    if (n < s) red[n] = fmaxf(red[n], red[n + s]);
    __syncthreads();
  }
  float mx = red[0];
  __syncthreads();
  float ex = __expf(v - mx);
  red[n] = ex;
  __syncthreads();
  for (int s = 128; s > 0; s >>= 1) {
    if (n < s) red[n] += red[n + s];
    __syncthreads();
  }
  out[(long long)row * 256 + n] = (__bf16)(ex / red[0]);
}

// ---------------------------------------------------------------------------
extern "C" void kernel_launch(void* const* d_in, const int* in_sizes, int n_in,
                              void* d_out, int out_size, void* d_ws, size_t ws_size,
                              hipStream_t stream) {
  (void)in_sizes; (void)n_in; (void)out_size; (void)ws_size;

  const float* e      = (const float*)d_in[0];   // [16,256,64,64]
  const float* pam    = (const float*)d_in[1];   // [16,256,16,16]
  const float* q_w    = (const float*)d_in[2];   // [256,256,3,3]
  const float* k_w    = (const float*)d_in[3];
  const float* v_w    = (const float*)d_in[4];
  const float* proj_w = (const float*)d_in[5];   // [256,256,1,1]
  float* out = (float*)d_out;                    // [16,256,64,64]

  char* ws = (char*)d_ws;
  size_t off = 0;
  auto take = [&](size_t bytes) -> char* {
    char* p = ws + off;
    off = (off + bytes + 255) & ~(size_t)255;
    return p;
  };
  const long long NBIG = 16LL * 256 * 4096;   // 16.78M
  const long long NSM  = 16LL * 256 * 256;    // 1.05M

  __bf16* x_bf    = (__bf16*)take(NBIG * 2);        // e in bf16
  __bf16* v_bf    = (__bf16*)take(NBIG * 2);        // v conv output
  __bf16* pool_bf = (__bf16*)take(NSM * 2);
  __bf16* pam_bf  = (__bf16*)take(NSM * 2);
  __bf16* qwt     = (__bf16*)take(9LL * 65536 * 2);
  __bf16* kwt     = (__bf16*)take(9LL * 65536 * 2);
  __bf16* vwt     = (__bf16*)take(9LL * 65536 * 2);
  __bf16* q_c     = (__bf16*)take(NSM * 2);
  __bf16* k_c     = (__bf16*)take(NSM * 2);
  __bf16* qn      = (__bf16*)take(NSM * 2);
  __bf16* knT     = (__bf16*)take(NSM * 2);
  float*  attnF   = (float* )take(NSM * 4);
  __bf16* attnB   = (__bf16*)take(NSM * 2);
  __bf16* projB   = (__bf16*)take(65536LL * 2);
  __bf16* m2      = (__bf16*)take(NSM * 2);         // proj @ softmax(attn)

  // Stage 0: casts / pool / weight tap-transpose
  cvt_bf16_kernel<<<(unsigned)((NBIG + 255) / 256), 256, 0, stream>>>(e, x_bf, NBIG);
  cvt_bf16_kernel<<<(unsigned)((NSM + 255) / 256), 256, 0, stream>>>(pam, pam_bf, NSM);
  cvt_bf16_kernel<<<256, 256, 0, stream>>>(proj_w, projB, 65536LL);
  pool_kernel<<<(unsigned)(NSM / 256), 256, 0, stream>>>(e, pool_bf);
  wtrans_kernel<<<2304, 256, 0, stream>>>(q_w, qwt);
  wtrans_kernel<<<2304, 256, 0, stream>>>(k_w, kwt);
  wtrans_kernel<<<2304, 256, 0, stream>>>(v_w, vwt);

  // Stage 1: convs as WMMA GEMMs
  // v = conv3x3(e):           M=256, N=4096, K=256 x 9 taps
  gemm_conv_wmma<<<dim3(2, 32, 16), 128, 0, stream>>>(
      vwt, 0, x_bf, 256LL * 4096, nullptr, v_bf, 256LL * 4096,
      nullptr, 0, 256, 4096, 256, 9, 6, 64, 64, 1.0f);
  // q = conv3x3(pool(e)):     M=256, N=256
  gemm_conv_wmma<<<dim3(2, 2, 16), 128, 0, stream>>>(
      qwt, 0, pool_bf, 65536LL, nullptr, q_c, 65536LL,
      nullptr, 0, 256, 256, 256, 9, 4, 16, 16, 1.0f);
  // k = conv3x3(pam_out)
  gemm_conv_wmma<<<dim3(2, 2, 16), 128, 0, stream>>>(
      kwt, 0, pam_bf, 65536LL, nullptr, k_c, 65536LL,
      nullptr, 0, 256, 256, 256, 9, 4, 16, 16, 1.0f);

  // Stage 2: L2 normalize (k transposed for the attention GEMM)
  l2norm_kernel<<<4096, 256, 0, stream>>>(q_c, qn, 0);
  l2norm_kernel<<<4096, 256, 0, stream>>>(k_c, knT, 1);

  // Stage 3: attn_raw = (qn . kn^T) / sqrt(256)
  gemm_conv_wmma<<<dim3(2, 2, 16), 128, 0, stream>>>(
      qn, 65536LL, knT, 65536LL, attnF, nullptr, 65536LL,
      nullptr, 0, 256, 256, 256, 1, 4, 16, 16, 0.0625f);
  softmax_kernel<<<4096, 256, 0, stream>>>(attnF, attnB);

  // Stage 4: fold proj into attention: m2 = proj_w @ softmax(attn)
  gemm_conv_wmma<<<dim3(2, 2, 16), 128, 0, stream>>>(
      projB, 0, attnB, 65536LL, nullptr, m2, 65536LL,
      nullptr, 0, 256, 256, 256, 1, 4, 16, 16, 1.0f);

  // Stage 5: out = m2 @ v + e   (f32 output, residual add)
  gemm_conv_wmma<<<dim3(2, 32, 16), 128, 0, stream>>>(
      m2, 65536LL, v_bf, 256LL * 4096, out, nullptr, 256LL * 4096,
      e, 256LL * 4096, 256, 4096, 256, 1, 6, 64, 64, 1.0f);
}
